// QuickMergePP_58815282152180
// MI455X (gfx1250) — matrix operand order repaired and checked by reading, not compiled, not run on hardware
//
#include <hip/hip_runtime.h>
#include <hip/hip_bf16.h>
#include <stdint.h>

// ---------------------------------------------------------------------------
// QuickMerge++ for MI455X (gfx1250, wave32, WMMA f16->f32, async LDS staging)
// B=4, N=4096, D=1024, H=256, K=512
// ---------------------------------------------------------------------------

typedef __attribute__((ext_vector_type(16))) _Float16 v16h;
typedef __attribute__((ext_vector_type(8)))  _Float16 v8h;
typedef __attribute__((ext_vector_type(4)))  _Float16 v4h;
typedef __attribute__((ext_vector_type(8)))  float    v8f;

#define B_  4
#define N_  4096
#define D_  1024
#define H_  256
#define K_  512
#define RSQRT_D 0.03125f          // 1/sqrt(1024)
#define RPAD_   2048              // rows per direction, padded (B*(K-1)=2044 -> 2048)

// ---------------- WMMA helpers -------------------------------------------
// A 16x32 f16 slice: lane l (l16=l&15 owns row M=l16, kh=l>>4):
//   elems 0..7  <- K = kh*8 + 0..7        (contiguous 16B)
//   elems 8..15 <- K = 16 + kh*8 + 0..7   (contiguous 16B)
__device__ __forceinline__ v16h load_a_slice(const _Float16* rowp, int kh) {
  v8h lo = *(const v8h*)(rowp + kh * 8);
  v8h hi = *(const v8h*)(rowp + 16 + kh * 8);
  v16h r;
#pragma unroll
  for (int i = 0; i < 8; ++i) { r[i] = lo[i]; r[i + 8] = hi[i]; }
  return r;
}
// B 32x16 f16 slice: lane owns column N=l16; elems 0..7 <- K = kh*8 + 0..7, elems 8..15 <- K = 16 + kh*8 + 0..7
__device__ __forceinline__ v16h load_b_slice(const _Float16* colp, int kh) {
  v8h lo = *(const v8h*)(colp + kh * 8);
  v8h hi = *(const v8h*)(colp + 16 + kh * 8);
  v16h r;
#pragma unroll
  for (int i = 0; i < 8; ++i) { r[i] = lo[i]; r[i + 8] = hi[i]; }
  return r;
}
__device__ __forceinline__ v8f wmma_f16(v16h a, v16h b, v8f c) {
  return __builtin_amdgcn_wmma_f32_16x16x32_f16(false, a, false, b,
                                                (short)0, c, false, false);
}
__device__ __forceinline__ v8f v8f_zero() {
  v8f z = {0.f, 0.f, 0.f, 0.f, 0.f, 0.f, 0.f, 0.f};
  return z;
}

// ---------------- CDNA5 async global->LDS staging -------------------------
// One instruction moves 16B per lane (512B per wave). 32KB tile = 8 insts/thread
// with 256 threads. ASYNCcnt tracks them; loads complete in order.
__device__ __forceinline__ void async_b128(uint32_t lds_addr, const _Float16* gsrc) {
  asm volatile("global_load_async_to_lds_b128 %0, %1, off"
               :: "v"(lds_addr), "v"(gsrc)
               : "memory");
}
__device__ __forceinline__ void async_tile_load_32k(const _Float16* gsrc,
                                                    uint32_t lds_base) {
#pragma unroll
  for (int j = 0; j < 8; ++j) {
    int ch = (int)threadIdx.x + 256 * j;          // 16B chunk index (0..2047)
    async_b128(lds_base + (uint32_t)ch * 16u, gsrc + ch * 8);
  }
}
__device__ __forceinline__ void wait_async_le8() {
  asm volatile("s_wait_asynccnt 0x8" ::: "memory");
}
__device__ __forceinline__ void wait_async_0() {
  asm volatile("s_wait_asynccnt 0x0" ::: "memory");
}

// ---------------- generic block reduce (power-of-2 blockDim) --------------
__device__ __forceinline__ float blockReduceSum(float v, float* red) {
  red[threadIdx.x] = v; __syncthreads();
  for (int s = blockDim.x >> 1; s > 0; s >>= 1) {
    if ((int)threadIdx.x < s) red[threadIdx.x] += red[threadIdx.x + s];
    __syncthreads();
  }
  float r = red[0]; __syncthreads();
  return r;
}
__device__ __forceinline__ float blockReduceMax(float v, float* red) {
  red[threadIdx.x] = v; __syncthreads();
  for (int s = blockDim.x >> 1; s > 0; s >>= 1) {
    if ((int)threadIdx.x < s) red[threadIdx.x] = fmaxf(red[threadIdx.x], red[threadIdx.x + s]);
    __syncthreads();
  }
  float r = red[0]; __syncthreads();
  return r;
}
__device__ __forceinline__ float blockReduceMin(float v, float* red) {
  red[threadIdx.x] = v; __syncthreads();
  for (int s = blockDim.x >> 1; s > 0; s >>= 1) {
    if ((int)threadIdx.x < s) red[threadIdx.x] = fminf(red[threadIdx.x], red[threadIdx.x + s]);
    __syncthreads();
  }
  float r = red[0]; __syncthreads();
  return r;
}

// ---------------- misc kernels --------------------------------------------
__global__ void zero_f32_kernel(float* p, unsigned long long n) {
  unsigned long long i = (unsigned long long)blockIdx.x * blockDim.x + threadIdx.x;
  if (i < n) p[i] = 0.f;
}

// per-token: norm, f16 copy, f16 normalized copy.  grid=B*N, block=256
__global__ void prep_x_kernel(const float* __restrict__ X,
                              _Float16* __restrict__ Xh,
                              _Float16* __restrict__ Xnh,
                              float* __restrict__ tn) {
  __shared__ float red[256];
  int tok = blockIdx.x;
  const float4* xp = (const float4*)(X + (size_t)tok * D_);
  float4 v = xp[threadIdx.x];
  float ss = v.x * v.x + v.y * v.y + v.z * v.z + v.w * v.w;
  float tot = blockReduceSum(ss, red);
  float nrm = sqrtf(tot);
  float inv = 1.0f / (nrm + 1e-12f);
  if (threadIdx.x == 0) tn[tok] = nrm;
  v4h h, hn;
  h[0] = (_Float16)v.x; h[1] = (_Float16)v.y; h[2] = (_Float16)v.z; h[3] = (_Float16)v.w;
  hn[0] = (_Float16)(v.x * inv); hn[1] = (_Float16)(v.y * inv);
  hn[2] = (_Float16)(v.z * inv); hn[3] = (_Float16)(v.w * inv);
  ((v4h*)(Xh  + (size_t)tok * D_))[threadIdx.x] = h;
  ((v4h*)(Xnh + (size_t)tok * D_))[threadIdx.x] = hn;
}

// transpose + f32->f16: dst[c*R + r] = src[r*C + c]
__global__ void convt_kernel(const float* __restrict__ src, _Float16* __restrict__ dst,
                             int R, int C) {
  int i = blockIdx.x * blockDim.x + threadIdx.x;
  if (i >= R * C) return;
  int r = i / C, c = i % C;
  dst[(size_t)c * R + r] = (_Float16)src[i];
}

// ---------------- saliency (flash-entropy over X X^T) ----------------------
// grid = (N/128, B), block = 256 (8 waves, each wave a 16-row tile).
// B column tiles (16 tokens x 1024 f16 = 32KB) are DMA'd once per block into
// LDS via global_load_async_to_lds_b128 (double-buffered, 64KB LDS),
// synchronized with s_wait_asynccnt + workgroup barriers. WMMA reads the B
// operand from LDS (ds_load_b128); the A operand streams from L2.
__global__ void __launch_bounds__(256) saliency_kernel(
    const _Float16* __restrict__ Xh,
    const float* __restrict__ tn,
    float* __restrict__ entropy) {
  __shared__ __align__(16) _Float16 sB[2][16 * D_];   // 2 x 32KB
  int b = blockIdx.y;
  int wave = threadIdx.x >> 5;
  int lane = threadIdx.x & 31;
  int l16 = lane & 15, kh = lane >> 4;
  int rowbase = blockIdx.x * 128 + wave * 16;
  const _Float16* Xb = Xh + (size_t)b * N_ * D_;
  const _Float16* Arow = Xb + (size_t)(rowbase + l16) * D_;

  float shiftv[8], Z[8], S[8];
#pragma unroll
  for (int r = 0; r < 8; ++r) {
    float t = tn[b * N_ + rowbase + r + 8 * kh];
    shiftv[r] = t * t * RSQRT_D;   // diagonal score = per-row softmax shift
    Z[r] = 0.f; S[r] = 0.f;
  }

  const int NT = N_ / 16;                               // 256 column tiles
  uint32_t lds0 = (uint32_t)(uintptr_t)&sB[0][0];
  uint32_t lds1 = (uint32_t)(uintptr_t)&sB[1][0];
  async_tile_load_32k(Xb, lds0);                        // prefetch tile 0

  for (int mt = 0; mt < NT; ++mt) {
    int buf = mt & 1;
    if (mt + 1 < NT) {
      async_tile_load_32k(Xb + (size_t)(mt + 1) * 16 * D_, buf ? lds0 : lds1);
      wait_async_le8();      // own tile-mt loads complete (in-order)
    } else {
      wait_async_0();
    }
    __syncthreads();         // all waves' tile-mt data visible in LDS

    const _Float16* Bcol = &sB[buf][l16 * D_];
    v8f acc = v8f_zero();
#pragma unroll 4
    for (int ks = 0; ks < D_ / 32; ++ks) {
      v16h a  = load_a_slice(Arow + ks * 32, kh);
      v16h bm = load_b_slice(Bcol + ks * 32, kh);
      acc = wmma_f16(a, bm, acc);
    }
#pragma unroll
    for (int r = 0; r < 8; ++r) {
      float s = acc[r] * RSQRT_D;
      float e = __expf(s - shiftv[r]);
      Z[r] += e; S[r] += s * e;
    }
    __syncthreads();         // everyone done reading buf before it is refilled
  }
  // reduce across the 16 lanes of each half-wave (xor <= 8 stays in-half)
#pragma unroll
  for (int r = 0; r < 8; ++r) {
#pragma unroll
    for (int m = 8; m >= 1; m >>= 1) {
      Z[r] += __shfl_xor(Z[r], m, 32);
      S[r] += __shfl_xor(S[r], m, 32);
    }
  }
  if (l16 == 0) {
#pragma unroll
    for (int r = 0; r < 8; ++r) {
      float Hh = shiftv[r] + __logf(Z[r]) - S[r] / Z[r];
      entropy[b * N_ + rowbase + r + 8 * kh] = Hh;
    }
  }
}

// ---------------- saliency -> gumbel softmax mask -> sal_mass --------------
__device__ __forceinline__ float gumbel_hash(uint32_t x) {
  x = x * 2654435761u + 42u;
  x ^= x >> 17; x *= 0xed5ad4bbu;
  x ^= x >> 11; x *= 0xac4c1b51u;
  x ^= x >> 15; x *= 0x31848babu;
  x ^= x >> 14;
  float u = ((float)(x >> 8) + 0.5f) * (1.0f / 16777216.0f);  // (0,1)
  return -__logf(-__logf(u));
}

// grid = B, block = 1024
__global__ void sal_mass_kernel(const float* __restrict__ entropy,
                                float* __restrict__ sal_mass) {
  __shared__ float red[1024];
  int b = blockIdx.x;
  const float* Hp = entropy + b * N_;
  float h[4];
#pragma unroll
  for (int i = 0; i < 4; ++i) h[i] = Hp[threadIdx.x + i * 1024];
  float lo = blockReduceMin(fminf(fminf(h[0], h[1]), fminf(h[2], h[3])), red);
  float hi = blockReduceMax(fmaxf(fmaxf(h[0], h[1]), fmaxf(h[2], h[3])), red);
  float sal[4], y[4];
#pragma unroll
  for (int i = 0; i < 4; ++i) {
    int n = threadIdx.x + i * 1024;
    sal[i] = 1.f - (h[i] - lo) / (hi - lo + 1e-8f);
    y[i] = (sal[i] + gumbel_hash((uint32_t)(b * N_ + n))) * 10.f;  // /TEMPERATURE
  }
  float ym = blockReduceMax(fmaxf(fmaxf(y[0], y[1]), fmaxf(y[2], y[3])), red);
  float e[4], es = 0.f;
#pragma unroll
  for (int i = 0; i < 4; ++i) { e[i] = __expf(y[i] - ym); es += e[i]; }
  float Zs = blockReduceSum(es, red);
#pragma unroll
  for (int i = 0; i < 4; ++i) {
    int n = threadIdx.x + i * 1024;
    float mask = e[i] / Zs;
    sal_mass[b * N_ + n] = mask * sal[i] + (1.f - mask) * 0.01f;  // EPSILON
  }
}

// ---------------- per-batch bitonic sort (desc by value, tie: low idx) -----
// grid = B, block = 1024. Optionally emits sorted values and/or top-k indices.
__global__ void bitonic_sort_kernel(const float* __restrict__ vals,
                                    float* __restrict__ outVals,
                                    int* __restrict__ outIdx, int outK) {
  __shared__ float sv[N_];
  __shared__ int   si[N_];
  int b = blockIdx.x;
  const float* v = vals + b * N_;
  for (int i = threadIdx.x; i < N_; i += blockDim.x) { sv[i] = v[i]; si[i] = i; }
  __syncthreads();
  for (int k = 2; k <= N_; k <<= 1) {
    for (int j = k >> 1; j > 0; j >>= 1) {
      for (int i = threadIdx.x; i < N_; i += blockDim.x) {
        int l = i ^ j;
        if (l > i) {
          bool up = ((i & k) == 0);
          bool firstWorse = (sv[i] < sv[l]) || (sv[i] == sv[l] && si[i] > si[l]);
          if (up ? firstWorse : !firstWorse) {
            float tv = sv[i]; sv[i] = sv[l]; sv[l] = tv;
            int   ti = si[i]; si[i] = si[l]; si[l] = ti;
          }
        }
      }
      __syncthreads();
    }
  }
  if (outVals)
    for (int i = threadIdx.x; i < N_; i += blockDim.x) outVals[b * N_ + i] = sv[i];
  if (outIdx)
    for (int i = threadIdx.x; i < outK; i += blockDim.x) outIdx[b * outK + i] = si[i];
}

// gather centroid rows (normalized f16). grid = (K, B), block = 256
__global__ void gather_cent_kernel(const _Float16* __restrict__ Xnh,
                                   const int* __restrict__ cidx,
                                   _Float16* __restrict__ cent) {
  int b = blockIdx.y, k = blockIdx.x;
  int src = cidx[b * K_ + k];
  const v4h* s = (const v4h*)(Xnh + ((size_t)(b * N_ + src)) * D_);
  v4h* d = (v4h*)(cent + ((size_t)(b * K_ + k)) * D_);
  d[threadIdx.x] = s[threadIdx.x];
}

// ---------------- sim = Xn @ cent^T, fused argmax -> labels ----------------
// grid = (N/128, B), block = 256
__global__ void sim_argmax_kernel(const _Float16* __restrict__ Xnh,
                                  const _Float16* __restrict__ cent,
                                  int* __restrict__ labels) {
  int b = blockIdx.y;
  int wave = threadIdx.x >> 5;
  int lane = threadIdx.x & 31;
  int l16 = lane & 15, kh = lane >> 4;
  int rowbase = blockIdx.x * 128 + wave * 16;
  const _Float16* Arow = Xnh + ((size_t)(b * N_ + rowbase + l16)) * D_;
  float best[8]; int bidx[8];
#pragma unroll
  for (int r = 0; r < 8; ++r) { best[r] = -3.4e38f; bidx[r] = 0x7fffffff; }
  for (int ct = 0; ct < K_ / 16; ++ct) {
    const _Float16* Bcol = cent + ((size_t)(b * K_ + ct * 16 + l16)) * D_;
    v8f acc = v8f_zero();
#pragma unroll 4
    for (int ks = 0; ks < D_ / 32; ++ks) {
      v16h a  = load_a_slice(Arow + ks * 32, kh);
      v16h bm = load_b_slice(Bcol + ks * 32, kh);
      acc = wmma_f16(a, bm, acc);
    }
    int col = ct * 16 + l16;
#pragma unroll
    for (int r = 0; r < 8; ++r) {
      float v = acc[r];
      if (v > best[r] || (v == best[r] && col < bidx[r])) { best[r] = v; bidx[r] = col; }
    }
  }
#pragma unroll
  for (int r = 0; r < 8; ++r) {
#pragma unroll
    for (int m = 8; m >= 1; m >>= 1) {
      float ov = __shfl_xor(best[r], m, 32);
      int   oi = __shfl_xor(bidx[r], m, 32);
      if (ov > best[r] || (ov == best[r] && oi < bidx[r])) { best[r] = ov; bidx[r] = oi; }
    }
  }
  if (l16 == 0) {
#pragma unroll
    for (int r = 0; r < 8; ++r)
      labels[b * N_ + rowbase + r + 8 * kh] = bidx[r];
  }
}

// group mass accumulation.  grid = B*N/256, block = 256
__global__ void group_mass_kernel(const int* __restrict__ labels,
                                  const float* __restrict__ sal_mass,
                                  float* __restrict__ gm) {
  int i = blockIdx.x * 256 + threadIdx.x;
  int b = i >> 12;
  atomicAdd(&gm[b * K_ + labels[i]], sal_mass[i]);
}

// merged[b,label,:] += w * X[b,n,:].  grid = B*N, block = 256
__global__ void scatter_merge_kernel(const float* __restrict__ X,
                                     const int* __restrict__ labels,
                                     const float* __restrict__ sal_mass,
                                     const float* __restrict__ gm,
                                     float* __restrict__ merged) {
  int tok = blockIdx.x;
  int b = tok >> 12;
  int lab = labels[tok];
  float w = sal_mass[tok] / (gm[b * K_ + lab] + 1e-8f);
  float4 v = ((const float4*)(X + (size_t)tok * D_))[threadIdx.x];
  float* mp = merged + ((size_t)(b * K_ + lab)) * D_;
  int d = threadIdx.x * 4;
  atomicAdd(&mp[d + 0], w * v.x);
  atomicAdd(&mp[d + 1], w * v.y);
  atomicAdd(&mp[d + 2], w * v.z);
  atomicAdd(&mp[d + 3], w * v.w);
}

// ---------------- fidelity -------------------------------------------------
// thr[b] = (1 - gamma)^2 * fro2, gamma = sum(top-512 tn)/sum(tn). grid=B, block=1024
__global__ void thr_kernel(const float* __restrict__ tn,
                           const float* __restrict__ sortedTn,
                           float* __restrict__ thr) {
  __shared__ float red[1024];
  int b = blockIdx.x;
  float sAll = 0.f, sSq = 0.f, sTop = 0.f;
  for (int i = threadIdx.x; i < N_; i += 1024) {
    float t = tn[b * N_ + i];
    sAll += t; sSq += t * t;
  }
  for (int i = threadIdx.x; i < K_; i += 1024) sTop += sortedTn[b * N_ + i];
  sAll = blockReduceSum(sAll, red);
  sSq  = blockReduceSum(sSq,  red);
  sTop = blockReduceSum(sTop, red);
  if (threadIdx.x == 0) {
    float gamma = sTop / sAll;
    float g1 = 1.f - gamma;
    thr[b] = g1 * g1 * sSq;
  }
}

// rec + relu hinge accumulation. grid = B*N/8, block = 256 (8 waves, 1 token/wave)
__global__ void rec_fid_kernel(const float* __restrict__ X,
                               const float* __restrict__ merged,
                               const float* __restrict__ thr,
                               float* __restrict__ fidAcc) {
  __shared__ float red[8];
  int wave = threadIdx.x >> 5, lane = threadIdx.x & 31;
  int tok = blockIdx.x * 8 + wave;
  int b = tok >> 12;
  int n = tok & (N_ - 1);
  const float* xp = X + (size_t)tok * D_;
  const float* mp = (n < K_) ? (merged + ((size_t)(b * K_ + n)) * D_) : nullptr;
  float ss = 0.f;
  for (int i = lane; i < D_; i += 32) {
    float d = xp[i] - (mp ? mp[i] : 0.f);
    ss += d * d;
  }
#pragma unroll
  for (int m = 16; m >= 1; m >>= 1) ss += __shfl_xor(ss, m, 32);
  if (lane == 0) red[wave] = fmaxf(ss * (1.0f / D_) - thr[b], 0.f);
  __syncthreads();
  if (threadIdx.x == 0) {
    float s = 0.f;
#pragma unroll
    for (int w = 0; w < 8; ++w) s += red[w];
    atomicAdd(fidAcc, s);
  }
}

// ---------------- AR loss --------------------------------------------------
// build forward/backward running-mean contexts as f16. grid = B*D/256, block = 256
__global__ void cumsum_ctx_kernel(const float* __restrict__ merged,
                                  _Float16* __restrict__ ctx) {
  int gid = blockIdx.x * 256 + threadIdx.x;       // 0..4095
  int b = gid >> 10, d = gid & (D_ - 1);
  const float* Mb = merged + ((size_t)(b * K_)) * D_ + d;
  _Float16* cf = ctx + d;                          // dir 0
  _Float16* cb = ctx + (size_t)RPAD_ * D_ + d;     // dir 1
  float cs = 0.f;
  for (int t = 0; t < K_ - 1; ++t) {
    cs += Mb[(size_t)t * D_];
    cf[(size_t)(b * (K_ - 1) + t) * D_] = (_Float16)(cs / (float)(t + 1));
  }
  float rs = 0.f;
  for (int t = K_ - 1; t >= 1; --t) {
    rs += Mb[(size_t)t * D_];
    cb[(size_t)(b * (K_ - 1) + (t - 1)) * D_] = (_Float16)(rs / (float)(K_ - t));
  }
}

// hidden = relu(ctx @ W1 + b1).  grid = (RPAD/128, H/16, 2), block = 256
__global__ void gemm1_kernel(const _Float16* __restrict__ ctx,
                             const _Float16* __restrict__ w1t,
                             const float* __restrict__ fb1,
                             const float* __restrict__ bb1,
                             _Float16* __restrict__ hidden) {
  int dir = blockIdx.z;
  int wave = threadIdx.x >> 5, lane = threadIdx.x & 31;
  int l16 = lane & 15, kh = lane >> 4;
  int rowbase = (blockIdx.x * 8 + wave) * 16;
  int hbase = blockIdx.y * 16;
  const _Float16* A  = ctx + ((size_t)dir * RPAD_ + rowbase + l16) * D_;
  const _Float16* Bp = w1t + ((size_t)dir * H_ + hbase + l16) * D_;
  const float* b1 = dir ? bb1 : fb1;
  v8f acc = v8f_zero();
#pragma unroll 4
  for (int ks = 0; ks < D_ / 32; ++ks) {
    v16h a  = load_a_slice(A  + ks * 32, kh);
    v16h bm = load_b_slice(Bp + ks * 32, kh);
    acc = wmma_f16(a, bm, acc);
  }
  int hcol = hbase + l16;
  float bias = b1[hcol];
#pragma unroll
  for (int r = 0; r < 8; ++r) {
    float v = fmaxf(acc[r] + bias, 0.f);
    hidden[((size_t)dir * RPAD_ + rowbase + r + 8 * kh) * H_ + hcol] = (_Float16)v;
  }
}

// pred = hidden @ W2 + b2, fused MSE vs targets. grid = (RPAD/128, D/16, 2), block=256
__global__ void gemm2_loss_kernel(const _Float16* __restrict__ hidden,
                                  const _Float16* __restrict__ w2t,
                                  const float* __restrict__ fb2,
                                  const float* __restrict__ bb2,
                                  const float* __restrict__ merged,
                                  float* __restrict__ lossAcc) {
  __shared__ float red[256];
  int dir = blockIdx.z;
  int wave = threadIdx.x >> 5, lane = threadIdx.x & 31;
  int l16 = lane & 15, kh = lane >> 4;
  int rowbase = (blockIdx.x * 8 + wave) * 16;
  int dbase = blockIdx.y * 16;
  const _Float16* A  = hidden + ((size_t)dir * RPAD_ + rowbase + l16) * H_;
  const _Float16* Bp = w2t + ((size_t)dir * D_ + dbase + l16) * H_;
  const float* b2 = dir ? bb2 : fb2;
  v8f acc = v8f_zero();
#pragma unroll
  for (int ks = 0; ks < H_ / 32; ++ks) {
    v16h a  = load_a_slice(A  + ks * 32, kh);
    v16h bm = load_b_slice(Bp + ks * 32, kh);
    acc = wmma_f16(a, bm, acc);
  }
  int dcol = dbase + l16;
  float bias = b2[dcol];
  float ls = 0.f;
#pragma unroll
  for (int r = 0; r < 8; ++r) {
    int g = rowbase + r + 8 * kh;       // global padded row
    int b = g / (K_ - 1);
    int t = g - b * (K_ - 1);
    if (b < B_) {                        // rows >= 2044 are padding
      float pred = acc[r] + bias;
      int trow = (dir == 0) ? (t + 1) : t;
      float tgt = merged[((size_t)(b * K_ + trow)) * D_ + dcol];
      float d = pred - tgt;
      ls += d * d;
    }
  }
  ls = blockReduceSum(ls, red);
  if (threadIdx.x == 0) atomicAdd(&lossAcc[dir], ls);
}

__global__ void finalize_kernel(const float* __restrict__ acc, float* __restrict__ out3) {
  float ar  = (acc[0] + acc[1]) * (1.0f / (float)(B_ * (K_ - 1) * D_));
  float fid = acc[2] * (1.0f / (float)(B_ * N_));
  out3[0] = ar;
  out3[1] = fid;
  out3[2] = 0.1f * ar + fid;   // ALPHA*ar + fid
}

// ---------------------------------------------------------------------------
extern "C" void kernel_launch(void* const* d_in, const int* in_sizes, int n_in,
                              void* d_out, int out_size, void* d_ws, size_t ws_size,
                              hipStream_t stream) {
  (void)in_sizes; (void)n_in; (void)out_size; (void)ws_size;
  const float* X      = (const float*)d_in[0];
  const float* fw_W1  = (const float*)d_in[1];
  const float* fw_b1  = (const float*)d_in[2];
  const float* fw_W2  = (const float*)d_in[3];
  const float* fw_b2  = (const float*)d_in[4];
  const float* bw_W1  = (const float*)d_in[5];
  const float* bw_b1  = (const float*)d_in[6];
  const float* bw_W2  = (const float*)d_in[7];
  const float* bw_b2  = (const float*)d_in[8];
  float* out = (float*)d_out;         // [B*K*D merged][ar][fid][total]
  float* merged = out;

  // -------- workspace carve-up (~80 MB) --------
  char* w = (char*)d_ws;
  auto alloc = [&](size_t bytes) -> char* {
    char* p = w; w += (bytes + 255) & ~(size_t)255; return p;
  };
  _Float16* Xh       = (_Float16*)alloc((size_t)B_ * N_ * D_ * 2);
  _Float16* Xnh      = (_Float16*)alloc((size_t)B_ * N_ * D_ * 2);
  float*    tn       = (float*)   alloc((size_t)B_ * N_ * 4);
  float*    entropy  = (float*)   alloc((size_t)B_ * N_ * 4);
  float*    sal_mass = (float*)   alloc((size_t)B_ * N_ * 4);
  int*      cidx     = (int*)     alloc((size_t)B_ * K_ * 4);
  _Float16* cent     = (_Float16*)alloc((size_t)B_ * K_ * D_ * 2);
  int*      labels   = (int*)     alloc((size_t)B_ * N_ * 4);
  float*    gm       = (float*)   alloc((size_t)B_ * K_ * 4);
  float*    sortedTn = (float*)   alloc((size_t)B_ * N_ * 4);
  _Float16* ctx      = (_Float16*)alloc((size_t)2 * RPAD_ * D_ * 2);
  _Float16* hidden   = (_Float16*)alloc((size_t)2 * RPAD_ * H_ * 2);
  _Float16* w1t      = (_Float16*)alloc((size_t)2 * H_ * D_ * 2);
  _Float16* w2t      = (_Float16*)alloc((size_t)2 * D_ * H_ * 2);
  float*    acc      = (float*)   alloc(256);   // [0]=loss_f [1]=loss_b [2]=fid
  float*    thr      = (float*)   alloc(256);

  const unsigned long long outN = (unsigned long long)B_ * K_ * D_ + 3ull;
  const unsigned long long ctxN = (unsigned long long)2 * RPAD_ * D_ / 2; // f16 as f32 words

  // -------- zero init --------
  zero_f32_kernel<<<(unsigned)((outN + 255) / 256), 256, 0, stream>>>(out, outN);
  zero_f32_kernel<<<1, 256, 0, stream>>>(acc, 8ull);
  zero_f32_kernel<<<(B_ * K_ + 255) / 256, 256, 0, stream>>>(gm, (unsigned long long)B_ * K_);
  zero_f32_kernel<<<(unsigned)((ctxN + 255) / 256), 256, 0, stream>>>((float*)ctx, ctxN);

  // -------- prep --------
  prep_x_kernel<<<B_ * N_, 256, 0, stream>>>(X, Xh, Xnh, tn);
  convt_kernel<<<(D_ * H_ + 255) / 256, 256, 0, stream>>>(fw_W1, w1t, D_, H_);
  convt_kernel<<<(D_ * H_ + 255) / 256, 256, 0, stream>>>(bw_W1, w1t + (size_t)H_ * D_, D_, H_);
  convt_kernel<<<(H_ * D_ + 255) / 256, 256, 0, stream>>>(fw_W2, w2t, H_, D_);
  convt_kernel<<<(H_ * D_ + 255) / 256, 256, 0, stream>>>(bw_W2, w2t + (size_t)D_ * H_, H_, D_);

  // -------- saliency / mask / mass --------
  saliency_kernel<<<dim3(N_ / 128, B_), 256, 0, stream>>>(Xh, tn, entropy);
  sal_mass_kernel<<<B_, 1024, 0, stream>>>(entropy, sal_mass);

  // -------- top-K + merge --------
  bitonic_sort_kernel<<<B_, 1024, 0, stream>>>(sal_mass, nullptr, cidx, K_);
  bitonic_sort_kernel<<<B_, 1024, 0, stream>>>(tn, sortedTn, nullptr, 0);
  gather_cent_kernel<<<dim3(K_, B_), 256, 0, stream>>>(Xnh, cidx, cent);
  sim_argmax_kernel<<<dim3(N_ / 128, B_), 256, 0, stream>>>(Xnh, cent, labels);
  group_mass_kernel<<<B_ * N_ / 256, 256, 0, stream>>>(labels, sal_mass, gm);
  scatter_merge_kernel<<<B_ * N_, 256, 0, stream>>>(X, labels, sal_mass, gm, merged);

  // -------- fidelity --------
  thr_kernel<<<B_, 1024, 0, stream>>>(tn, sortedTn, thr);
  rec_fid_kernel<<<B_ * N_ / 8, 256, 0, stream>>>(X, merged, thr, &acc[2]);

  // -------- AR loss --------
  cumsum_ctx_kernel<<<B_ * D_ / 256, 256, 0, stream>>>(merged, ctx);
  gemm1_kernel<<<dim3(RPAD_ / 128, H_ / 16, 2), 256, 0, stream>>>(ctx, w1t, fw_b1, bw_b1, hidden);
  gemm2_loss_kernel<<<dim3(RPAD_ / 128, D_ / 16, 2), 256, 0, stream>>>(hidden, w2t, fw_b2, bw_b2,
                                                                       merged, acc);
  finalize_kernel<<<1, 1, 0, stream>>>(acc, out + (size_t)B_ * K_ * D_);
}
